// LlamaAttention_85538568667279
// MI455X (gfx1250) — 16-bit and 8-bit matrix operand order repaired and checked by reading, compiled, not run on hardware
//
#include <hip/hip_runtime.h>
#include <hip/hip_bf16.h>

typedef int      v8i  __attribute__((ext_vector_type(8)));
typedef float    v8f  __attribute__((ext_vector_type(8)));
typedef _Float16 v16h __attribute__((ext_vector_type(16)));

#define T_TOK   2048
#define HID     4096
#define NH      32
#define NKV     8
#define HD      128
#define QKV_OUT ((NH + 2 * NKV) * HD)
#define SCALING 0.08838834764831845f

__global__ void __launch_bounds__(256)
pack_acts_kernel(const int* __restrict__ qa, signed char* __restrict__ a8,
                 float* __restrict__ a_sum) {
  const int t = blockIdx.x;
  const int* row = qa + (size_t)t * HID;
  signed char* out = a8 + (size_t)t * HID;
  int local = 0;
  for (int i = threadIdx.x; i < HID; i += blockDim.x) {
    int v = row[i];
    out[i] = (signed char)v;
    local += v;
  }
  __shared__ int red[256];
  red[threadIdx.x] = local;
  __syncthreads();
  for (int s = 128; s > 0; s >>= 1) {
    if (threadIdx.x < s) red[threadIdx.x] += red[threadIdx.x + s];
    __syncthreads();
  }
  if (threadIdx.x == 0) a_sum[t] = (float)red[0];
}

__global__ void __launch_bounds__(256)
pack_w_kernel(const int* __restrict__ w, unsigned char* __restrict__ w8, int n) {
  int i = blockIdx.x * blockDim.x + threadIdx.x;
  if (i < n) w8[i] = (unsigned char)w[i];
}

union FragA8 { v8i v; uint2 d2[4]; };
union FragB8 { v8i v; uint4 d4[2]; };

__global__ void __launch_bounds__(256)
gemm_w4a8_kernel(const signed char* __restrict__ A8,
                 const float* __restrict__ a_sum,
                 const float* __restrict__ s_a,
                 const unsigned char* __restrict__ W8,
                 const float* __restrict__ s_w,
                 const float* __restrict__ z_w,
                 float* __restrict__ out, int N) {
  const int wave = threadIdx.x >> 5;
  const int lane = threadIdx.x & 31;
  const int wm = wave >> 2;
  const int wn = wave & 3;
  const int lo = lane & 15;
  const int hi = lane >> 4;
  const int hi8  = hi * 8;
  const int hi16 = hi * 16;

  const int mblock = blockIdx.y * 64;
  const int mtw = mblock + wm * 32;
  const int ntw = blockIdx.x * 128 + wn * 32;

  __shared__ __align__(16) signed char Ash[2][64 * 64];

  const int arow = threadIdx.x >> 2;
  const int aseg = threadIdx.x & 3;
  const signed char* agbase = A8 + (size_t)(mblock + arow) * HID + aseg * 16;
  unsigned ldsA[2];
  ldsA[0] = (unsigned)(size_t)&Ash[0][arow * 64 + aseg * 16];
  ldsA[1] = (unsigned)(size_t)&Ash[1][arow * 64 + aseg * 16];

  const unsigned char* brow0 = W8 + (size_t)(ntw + lo) * HID;
  const unsigned char* brow1 = W8 + (size_t)(ntw + 16 + lo) * HID;

  v8i acc00 = {0, 0, 0, 0, 0, 0, 0, 0};
  v8i acc01 = acc00, acc10 = acc00, acc11 = acc00;

  {
    unsigned long long ga = (unsigned long long)agbase;
    asm volatile("global_load_async_to_lds_b128 %0, %1, off"
                 :: "v"(ldsA[0]), "v"(ga) : "memory");
  }

  const int KSTEPS = HID / 64;
  for (int step = 0; step < KSTEPS; ++step) {
    asm volatile("s_wait_asynccnt 0" ::: "memory");
    __syncthreads();
    if (step + 1 < KSTEPS) {
      unsigned long long ga = (unsigned long long)(agbase + (step + 1) * 64);
      asm volatile("global_load_async_to_lds_b128 %0, %1, off"
                   :: "v"(ldsA[(step + 1) & 1]), "v"(ga) : "memory");
    }

    const signed char* abuf = &Ash[step & 1][0];
    const int kb = step * 64;

    FragA8 a0, a1;
    {
      const signed char* ap0 = abuf + (wm * 32 + lo) * 64 + hi16;
      const signed char* ap1 = ap0 + 16 * 64;
#pragma unroll
      for (int g = 0; g < 4; ++g) {
        a0.d2[g] = *(const uint2*)(ap0 + (g >> 1) * 32 + (g & 1) * 8);
        a1.d2[g] = *(const uint2*)(ap1 + (g >> 1) * 32 + (g & 1) * 8);
      }
    }
    FragB8 b0, b1;
    b0.d4[0] = *(const uint4*)(brow0 + kb + hi16);
    b0.d4[1] = *(const uint4*)(brow0 + kb + 32 + hi16);
    b1.d4[0] = *(const uint4*)(brow1 + kb + hi16);
    b1.d4[1] = *(const uint4*)(brow1 + kb + 32 + hi16);

    acc00 = __builtin_amdgcn_wmma_i32_16x16x64_iu8(true, a0.v, false, b0.v,
                                                   acc00, false, false);
    acc01 = __builtin_amdgcn_wmma_i32_16x16x64_iu8(true, a0.v, false, b1.v,
                                                   acc01, false, false);
    acc10 = __builtin_amdgcn_wmma_i32_16x16x64_iu8(true, a1.v, false, b0.v,
                                                   acc10, false, false);
    acc11 = __builtin_amdgcn_wmma_i32_16x16x64_iu8(true, a1.v, false, b1.v,
                                                   acc11, false, false);
  }

  const int n0 = ntw + lo;
  const int n1 = ntw + 16 + lo;
  const float sw0 = s_w[n0], zw0 = z_w[n0];
  const float sw1 = s_w[n1], zw1 = z_w[n1];
#pragma unroll
  for (int v = 0; v < 8; ++v) {
    const int m0 = mtw + v + 8 * hi;
    const int m1 = m0 + 16;
    const float sa0 = s_a[m0], su0 = a_sum[m0];
    const float sa1 = s_a[m1], su1 = a_sum[m1];
    out[(size_t)m0 * N + n0] = sa0 * sw0 * ((float)acc00[v] - zw0 * su0);
    out[(size_t)m0 * N + n1] = sa0 * sw1 * ((float)acc01[v] - zw1 * su0);
    out[(size_t)m1 * N + n0] = sa1 * sw0 * ((float)acc10[v] - zw0 * su1);
    out[(size_t)m1 * N + n1] = sa1 * sw1 * ((float)acc11[v] - zw1 * su1);
  }
}

__global__ void __launch_bounds__(256)
rope_kernel(const float* __restrict__ qkv, const int* __restrict__ positions,
            _Float16* __restrict__ qh, _Float16* __restrict__ kh,
            _Float16* __restrict__ vT) {
  const int t = blockIdx.x;
  const float pos = (float)positions[t];
  const float* row = qkv + (size_t)t * QKV_OUT;
  const float LOG_THETA = 9.210340371976184f;

  for (int i = threadIdx.x; i < NH * 64; i += blockDim.x) {
    const int h = i >> 6, d = i & 63;
    const float invf = __expf(-(2.0f * (float)d / (float)HD) * LOG_THETA);
    const float fr = pos * invf;
    const float c = __cosf(fr), s = __sinf(fr);
    const float x1 = row[h * HD + d];
    const float x2 = row[h * HD + 64 + d];
    _Float16* dst = qh + ((size_t)h * T_TOK + t) * HD;
    dst[d]      = (_Float16)((x1 * c - x2 * s) * SCALING);
    dst[64 + d] = (_Float16)((x2 * c + x1 * s) * SCALING);
  }
  for (int i = threadIdx.x; i < NKV * 64; i += blockDim.x) {
    const int h = i >> 6, d = i & 63;
    const float invf = __expf(-(2.0f * (float)d / (float)HD) * LOG_THETA);
    const float fr = pos * invf;
    const float c = __cosf(fr), s = __sinf(fr);
    const float x1 = row[NH * HD + h * HD + d];
    const float x2 = row[NH * HD + h * HD + 64 + d];
    _Float16* dst = kh + ((size_t)h * T_TOK + t) * HD;
    dst[d]      = (_Float16)(x1 * c - x2 * s);
    dst[64 + d] = (_Float16)(x2 * c + x1 * s);
  }
  for (int i = threadIdx.x; i < NKV * HD; i += blockDim.x) {
    const int h = i >> 7, d = i & 127;
    vT[((size_t)h * HD + d) * T_TOK + t] =
        (_Float16)row[(NH + NKV) * HD + h * HD + d];
  }
}

union FragF16 { v16h v; uint4 q[2]; };

__global__ void __launch_bounds__(32)
attn_kernel(const _Float16* __restrict__ qh, const _Float16* __restrict__ kh,
            const _Float16* __restrict__ vT, float* __restrict__ attn) {
  const int lane = threadIdx.x;
  const int h  = blockIdx.y;
  const int q0 = blockIdx.x * 16;
  const int hk = h / (NH / NKV);
  const int hi = lane >> 4;
  const int lo = lane & 15;

  __shared__ __align__(16) _Float16 pSh[16 * 32];

  FragF16 qf[4];
  {
    const _Float16* qrow = qh + ((size_t)h * T_TOK + (q0 + lo)) * HD;
#pragma unroll
    for (int c = 0; c < 4; ++c) {
      qf[c].q[0] = *(const uint4*)(qrow + c * 32 + hi * 16);
      qf[c].q[1] = *(const uint4*)(qrow + c * 32 + hi * 16 + 8);
    }
  }

  float m[8], l[8];
  v8f o[8];
#pragma unroll
  for (int v = 0; v < 8; ++v) { m[v] = -1e30f; l[v] = 0.0f; }
#pragma unroll
  for (int n = 0; n < 8; ++n) o[n] = (v8f){0, 0, 0, 0, 0, 0, 0, 0};

  const int kend = ((q0 + 15) / 32) * 32;
  for (int k0 = 0; k0 <= kend; k0 += 32) {
    v8f s0 = (v8f){0, 0, 0, 0, 0, 0, 0, 0};
    v8f s1 = (v8f){0, 0, 0, 0, 0, 0, 0, 0};
    const _Float16* kb0p = kh + ((size_t)hk * T_TOK + (k0 + lo)) * HD;
    const _Float16* kb1p = kb0p + 16 * HD;
#pragma unroll
    for (int c = 0; c < 4; ++c) {
      FragF16 kb0, kb1;
      kb0.q[0] = *(const uint4*)(kb0p + c * 32 + hi * 16);
      kb0.q[1] = *(const uint4*)(kb0p + c * 32 + hi * 16 + 8);
      kb1.q[0] = *(const uint4*)(kb1p + c * 32 + hi * 16);
      kb1.q[1] = *(const uint4*)(kb1p + c * 32 + hi * 16 + 8);
      s0 = __builtin_amdgcn_wmma_f32_16x16x32_f16(false, qf[c].v, false, kb0.v,
                                                  (short)0, s0, false, false);
      s1 = __builtin_amdgcn_wmma_f32_16x16x32_f16(false, qf[c].v, false, kb1.v,
                                                  (short)0, s1, false, false);
    }

    float scale[8];
#pragma unroll
    for (int v = 0; v < 8; ++v) {
      const int row  = q0 + v + 8 * hi;
      const int key0 = k0 + lo;
      float a = (key0      <= row) ? s0[v] : -1e30f;
      float b = (key0 + 16 <= row) ? s1[v] : -1e30f;
      float mx = fmaxf(a, b);
      mx = fmaxf(mx, __shfl_xor(mx, 1));
      mx = fmaxf(mx, __shfl_xor(mx, 2));
      mx = fmaxf(mx, __shfl_xor(mx, 4));
      mx = fmaxf(mx, __shfl_xor(mx, 8));
      const float nm = fmaxf(m[v], mx);
      scale[v] = __expf(m[v] - nm);
      const float p0 = __expf(a - nm);
      const float p1 = __expf(b - nm);
      s0[v] = p0;
      s1[v] = p1;
      float ts = p0 + p1;
      ts += __shfl_xor(ts, 1);
      ts += __shfl_xor(ts, 2);
      ts += __shfl_xor(ts, 4);
      ts += __shfl_xor(ts, 8);
      l[v] = l[v] * scale[v] + ts;
      m[v] = nm;
    }
#pragma unroll
    for (int n = 0; n < 8; ++n)
#pragma unroll
      for (int v = 0; v < 8; ++v) o[n][v] *= scale[v];

#pragma unroll
    for (int v = 0; v < 8; ++v) {
      const int prow = v + 8 * hi;
      pSh[prow * 32 + lo]      = (_Float16)s0[v];
      pSh[prow * 32 + 16 + lo] = (_Float16)s1[v];
    }
    __syncthreads();
    FragF16 pf;
    pf.q[0] = *(const uint4*)(&pSh[lo * 32 + hi * 16]);
    pf.q[1] = *(const uint4*)(&pSh[lo * 32 + hi * 16 + 8]);
    __syncthreads();

    const _Float16* vb = vT + (size_t)hk * HD * T_TOK + k0 + hi * 16;
#pragma unroll
    for (int n = 0; n < 8; ++n) {
      FragF16 vf;
      const _Float16* vp = vb + (size_t)(n * 16 + lo) * T_TOK;
      vf.q[0] = *(const uint4*)(vp);
      vf.q[1] = *(const uint4*)(vp + 8);
      o[n] = __builtin_amdgcn_wmma_f32_16x16x32_f16(false, pf.v, false, vf.v,
                                                    (short)0, o[n], false, false);
    }
  }

#pragma unroll
  for (int v = 0; v < 8; ++v) {
    const float inv = 1.0f / l[v];
    const int row = q0 + v + 8 * hi;
#pragma unroll
    for (int n = 0; n < 8; ++n) {
      attn[(size_t)row * (NH * HD) + h * HD + n * 16 + lo] = o[n][v] * inv;
    }
  }
}

__global__ void __launch_bounds__(256)
dynquant_kernel(const float* __restrict__ attn, signed char* __restrict__ a8,
                float* __restrict__ s2, float* __restrict__ sum2) {
  const int t = blockIdx.x;
  const float* row = attn + (size_t)t * HID;
  __shared__ float red[256];

  float mx = 0.0f;
  for (int i = threadIdx.x; i < HID; i += blockDim.x) mx = fmaxf(mx, fabsf(row[i]));
  red[threadIdx.x] = mx;
  __syncthreads();
  for (int s = 128; s > 0; s >>= 1) {
    if (threadIdx.x < s) red[threadIdx.x] = fmaxf(red[threadIdx.x], red[threadIdx.x + s]);
    __syncthreads();
  }
  const float scale = fmaxf(red[0], 1e-8f) / 127.0f;
  __syncthreads();

  float local = 0.0f;
  signed char* out = a8 + (size_t)t * HID;
  for (int i = threadIdx.x; i < HID; i += blockDim.x) {
    float q = rintf(row[i] / scale);
    q = fminf(fmaxf(q, -127.0f), 127.0f);
    out[i] = (signed char)q;
    local += q;
  }
  red[threadIdx.x] = local;
  __syncthreads();
  for (int s = 128; s > 0; s >>= 1) {
    if (threadIdx.x < s) red[threadIdx.x] += red[threadIdx.x + s];
    __syncthreads();
  }
  if (threadIdx.x == 0) {
    s2[t] = scale;
    sum2[t] = red[0];
  }
}

extern "C" void kernel_launch(void* const* d_in, const int* in_sizes, int n_in,
                              void* d_out, int out_size, void* d_ws, size_t ws_size,
                              hipStream_t stream) {
  const int*   positions   = (const int*)d_in[0];
  const int*   q_act       = (const int*)d_in[1];
  const float* act_scale   = (const float*)d_in[2];
  const int*   w_qkv_q     = (const int*)d_in[3];
  const float* w_qkv_scale = (const float*)d_in[4];
  const float* w_qkv_zero  = (const float*)d_in[5];
  const int*   w_o_q       = (const int*)d_in[6];
  const float* w_o_scale   = (const float*)d_in[7];
  const float* w_o_zero    = (const float*)d_in[8];
  float* out = (float*)d_out;

  char* ws = (char*)d_ws;
  size_t off = 0;
  auto alloc = [&](size_t bytes) -> void* {
    void* p = ws + off;
    off = (off + bytes + 255) & ~(size_t)255;
    return p;
  };

  signed char*   a8_act = (signed char*)  alloc((size_t)T_TOK * HID);
  float*         a_sum  = (float*)        alloc((size_t)T_TOK * 4);
  unsigned char* wq8    = (unsigned char*)alloc((size_t)QKV_OUT * HID);
  unsigned char* wo8    = (unsigned char*)alloc((size_t)HID * HID);
  float*         qkv    = (float*)        alloc((size_t)T_TOK * QKV_OUT * 4);
  _Float16*      q_h    = (_Float16*)     alloc((size_t)NH * T_TOK * HD * 2);
  _Float16*      k_h    = (_Float16*)     alloc((size_t)NKV * T_TOK * HD * 2);
  _Float16*      v_T    = (_Float16*)     alloc((size_t)NKV * HD * T_TOK * 2);
  float*         attn   = (float*)        alloc((size_t)T_TOK * HID * 4);
  signed char*   a8_2   = (signed char*)  alloc((size_t)T_TOK * HID);
  float*         s2     = (float*)        alloc((size_t)T_TOK * 4);
  float*         sum2   = (float*)        alloc((size_t)T_TOK * 4);
  (void)ws_size; (void)in_sizes; (void)n_in; (void)out_size;

  pack_acts_kernel<<<T_TOK, 256, 0, stream>>>(q_act, a8_act, a_sum);

  {
    int n1 = QKV_OUT * HID;
    pack_w_kernel<<<(n1 + 255) / 256, 256, 0, stream>>>(w_qkv_q, wq8, n1);
    int n2 = HID * HID;
    pack_w_kernel<<<(n2 + 255) / 256, 256, 0, stream>>>(w_o_q, wo8, n2);
  }

  gemm_w4a8_kernel<<<dim3(QKV_OUT / 128, T_TOK / 64), 256, 0, stream>>>(
      a8_act, a_sum, act_scale, wq8, w_qkv_scale, w_qkv_zero, qkv, QKV_OUT);

  rope_kernel<<<T_TOK, 256, 0, stream>>>(qkv, positions, q_h, k_h, v_T);

  attn_kernel<<<dim3(T_TOK / 16, NH), 32, 0, stream>>>(q_h, k_h, v_T, attn);

  dynquant_kernel<<<T_TOK, 256, 0, stream>>>(attn, a8_2, s2, sum2);

  gemm_w4a8_kernel<<<dim3(HID / 128, T_TOK / 64), 256, 0, stream>>>(
      a8_2, sum2, s2, wo8, w_o_scale, w_o_zero, out, HID);
}
